// SincNetLayer1D_50062138802217
// MI455X (gfx1250) — matrix operand order repaired and checked by reading, not compiled, not run on hardware
//
#include <hip/hip_runtime.h>
#include <stdint.h>

#define PI_F 3.14159265358979323846f
#define USE_ASYNC_STAGE 1

typedef __attribute__((ext_vector_type(16))) _Float16 v16h;
typedef __attribute__((ext_vector_type(8)))  _Float16 v8h;
typedef __attribute__((ext_vector_type(8)))  float    v8f;

static __device__ inline v8f wmma_f16(v16h a, v16h b, v8f c) {
  return __builtin_amdgcn_wmma_f32_16x16x32_f16(
      /*neg_a=*/false, a, /*neg_b=*/false, b,
      /*c_mod=*/(short)0, c, /*reuse_a=*/false, /*reuse_b=*/false);
}

static __device__ inline unsigned short h2u(_Float16 h) {
  return __builtin_bit_cast(unsigned short, h);
}

// Per-lane 16B global -> LDS async copy (GLOBAL_LOAD_ASYNC_TO_LDS_B128).
// lds_off = byte offset within workgroup LDS (low 32 bits of generic addr).
static __device__ inline void async_copy16(uint32_t lds_off, const void* gptr) {
  asm volatile("global_load_async_to_lds_b128 %0, %1, off"
               :: "v"(lds_off), "v"((uint64_t)(uintptr_t)gptr)
               : "memory");
}
static __device__ inline void wait_async0() {
  asm volatile("s_wait_asynccnt 0x0" ::: "memory");
}

// ---------------------------------------------------------------------------
// Kernel 1: Gaussian mask (stored as duplicated half2 per position) and
// sinc bandpass kr/ki (F x 129, f32). One block per filter.
// ---------------------------------------------------------------------------
__global__ __launch_bounds__(256) void sincnet_build_filters(
    const float* __restrict__ fc, const float* __restrict__ fbw,
    const float* __restrict__ tl, const float* __restrict__ st,
    uint32_t* __restrict__ maskh2, float* __restrict__ kr,
    float* __restrict__ ki) {
  const int f = blockIdx.x;
  const int tid = threadIdx.x;
  __shared__ float red[256];

  const float fcv = fc[f];
  const float band = fbw[f] * 16000.0f;
  const float tlv = tl[f];
  const float center = 0.5f * (st[f] + tlv);
  const float sd = tlv * (1.0f / 1.665f) + 1e-9f;

  // ---- Gaussian time mask over L=2048 ----
  float gv[8];
  float lmax = 0.0f;
  for (int k = 0; k < 8; ++k) {
    const int t = tid + 256 * k;
    const float z = ((float)t - center) / sd;
    const float g = expf(-0.5f * z * z);
    gv[k] = g;
    lmax = fmaxf(lmax, g);
  }
  red[tid] = lmax;
  __syncthreads();
  for (int s = 128; s > 0; s >>= 1) {
    if (tid < s) red[tid] = fmaxf(red[tid], red[tid + s]);
    __syncthreads();
  }
  const float gmax = red[0] + 1e-9f;
  __syncthreads();
  for (int k = 0; k < 8; ++k) {
    const unsigned u = h2u((_Float16)(gv[k] / gmax));
    maskh2[f * 2048 + tid + 256 * k] = u | (u << 16);
  }

  // ---- windowed sinc bandpass (fsize = 129) ----
  float bp = 0.0f;
  if (tid < 129) {
    if (tid == 64) {
      bp = 1.0f;
    } else {
      const int kk = (tid < 64) ? (63 - tid) : (tid - 65);
      const float tr = (float)(kk + 1) * (1.0f / 16000.0f);
      const float arg = 2.0f * PI_F * band * tr;
      bp = sinf(arg) / arg;
    }
    const float nj = (float)tid * (129.0f / 128.0f);
    const float win = 0.54f - 0.46f * cosf(2.0f * PI_F * nj / 129.0f);
    bp = bp * 2.0f * band * win;
  }
  red[tid] = (tid < 129) ? fabsf(bp) : 0.0f;
  __syncthreads();
  for (int s = 128; s > 0; s >>= 1) {
    if (tid < s) red[tid] = fmaxf(red[tid], red[tid + s]);
    __syncthreads();
  }
  const float bmax = red[0] + 1e-9f;
  if (tid < 129) {
    const float nj = (float)tid * (129.0f / 128.0f);
    const float ph = 2.0f * PI_F * fcv * nj;
    const float b = bp / bmax;
    kr[f * 129 + tid] = b * cosf(ph);
    ki[f * 129 + tid] = b * sinf(ph);
  }
}

// ---------------------------------------------------------------------------
// Kernel 2: convert input (B,L,2) f32 -> packed half2 per position.
// ---------------------------------------------------------------------------
__global__ __launch_bounds__(256) void sincnet_cvt_input(
    const float* __restrict__ in, uint32_t* __restrict__ inh) {
  const int i = blockIdx.x * 256 + threadIdx.x;  // 262144 positions
  const float2 v = ((const float2*)in)[i];
  const unsigned a = h2u((_Float16)v.x);
  const unsigned b = h2u((_Float16)v.y);
  inh[i] = a | (b << 16);
}

// ---------------------------------------------------------------------------
// Kernel 3: Toeplitz B blocks (K=288 x N=16 per filter per component) in the
// gfx1250 f16 B-matrix per-lane layout: elem kappa = 32q + 8h + j (+8 for j>=8).
// kappa = 2*pi + c ; B[kappa, n] = W_c[pi - n], tap valid 0..128.
// ---------------------------------------------------------------------------
__global__ __launch_bounds__(256) void sincnet_build_toeplitz(
    const float* __restrict__ kr, const float* __restrict__ ki,
    _Float16* __restrict__ Bt) {
  const int f = blockIdx.x;
  for (int idx = threadIdx.x; idx < 576; idx += 256) {
    const int comp = idx / 288;  // 0 = real out, 1 = imag out
    const int rem = idx % 288;
    const int q = rem >> 5;
    const int lane = rem & 31;
    const int n = lane & 15;
    const int h = lane >> 4;
    const size_t base = ((((size_t)f * 2 + comp) * 9 + q) * 32 + lane) * 16;
#pragma unroll
    for (int j = 0; j < 16; ++j) {
      const int kap = 32 * q + 8 * h + j + (j >= 8 ? 8 : 0);
      const int pi = kap >> 1;
      const int c = kap & 1;
      const int t = pi - n;
      float v = 0.0f;
      if (t >= 0 && t <= 128) {
        const float r = kr[f * 129 + t];
        const float im = ki[f * 129 + t];
        // out_real = conv(real,kr) - conv(imag,ki); out_imag = conv(real,ki) + conv(imag,kr)
        v = (comp == 0) ? (c ? -im : r) : (c ? r : im);
      }
      Bt[base + j] = (_Float16)v;
    }
  }
}

// ---------------------------------------------------------------------------
// Kernel 4: main WMMA conv. Block = (16 l) x (16 b) tile, 8 waves.
// in_lds row m holds halves [144 pos x 2 chan], stride 296 halves (592B).
// A fragment addresses are contiguous: half index = n*296 + 32q + 8h + j(+16).
// ---------------------------------------------------------------------------
__global__ __launch_bounds__(256) void sincnet_wmma_main(
    const uint32_t* __restrict__ inh, const uint32_t* __restrict__ maskh2,
    const _Float16* __restrict__ Bt, float* __restrict__ out) {
  __shared__ __attribute__((aligned(16))) _Float16 in_lds[16 * 296];
  __shared__ __attribute__((aligned(16))) uint32_t mask_lds[8 * 160];
  __shared__ __attribute__((aligned(16))) float out_lds[256 * 36];

  const int l0 = blockIdx.x * 16;
  const int b0 = blockIdx.y * 16;
  const int p0 = l0 - 64;
  const int tid = threadIdx.x;
  const int wave = tid >> 5;
  const int lane = tid & 31;
  const int n = lane & 15;  // A row (b) == B/D column (l)
  const int h = lane >> 4;

  // ---- stage input tile: async 16B DMA chunks for the in-range span ----
  const int pstart = p0 < 0 ? 0 : p0;
  const int pend = (p0 + 144) > 2048 ? 2048 : (p0 + 144);
  const int span = pend - pstart;  // multiple of 16 positions

  if (span != 144) {
    // zero only OOB cells (disjoint from async-written region)
    for (int e = tid; e < 16 * 148; e += 256) {
      const int row = e / 148;
      const int col = e - row * 148;  // one u32 == one position (2 halves)
      const int p = p0 + col;
      if (col >= 144 || p < 0 || p >= 2048)
        ((uint32_t*)in_lds)[row * 148 + col] = 0;
    }
  }
#if USE_ASYNC_STAGE
  {
    const uint32_t lbase = (uint32_t)(uintptr_t)in_lds;
    const int m = tid >> 4;  // 16 threads per row
    const uint32_t lrow = lbase + m * 592 + (pstart - p0) * 4;
    const uint32_t* grow = inh + ((size_t)(b0 + m) * 2048 + pstart);
    const int cpr = span >> 2;  // 16B chunks per row
    for (int wc = (tid & 15); wc < cpr; wc += 16)
      async_copy16(lrow + wc * 16, grow + wc * 4);
    wait_async0();
  }
#else
  for (int e = tid; e < 16 * 148; e += 256) {
    const int row = e / 148;
    const int col = e - row * 148;
    const int p = p0 + col;
    uint32_t v = 0;
    if (col < 144 && p >= 0 && p < 2048)
      v = inh[(size_t)(b0 + row) * 2048 + p];
    ((uint32_t*)in_lds)[row * 148 + col] = v;
  }
#endif
  __syncthreads();

  const _Float16* mask_h = (const _Float16*)mask_lds;

  for (int r = 0; r < 4; ++r) {
    for (int it = 0; it < 2; ++it) {
      const int f = 16 * r + wave * 2 + it;

      // per-wave mask strip: uniform 5 iterations, clamped duplicate tail
#pragma unroll
      for (int s5 = 0; s5 < 5; ++s5) {
        int s = lane + 32 * s5;          // 0..159
        s = s < 144 ? s : 143;
        int p = p0 + s;
        p = p < 0 ? 0 : (p > 2047 ? 2047 : p);
        mask_lds[wave * 160 + s] = maskh2[f * 2048 + p];
      }

      v8f cr = {};
      v8f ci = {};
      const size_t fr = (((size_t)f * 2 + 0) * 9 * 32 + lane) * 16;
      const size_t fi = (((size_t)f * 2 + 1) * 9 * 32 + lane) * 16;
      const int abase = n * 296 + 8 * h;          // halves
      const int mbase = wave * 320 + 8 * h;       // halves (dup half2)

      for (int q = 0; q < 9; ++q) {
        const v8h i0 = *(const v8h*)(in_lds + abase + 32 * q);
        const v8h i1 = *(const v8h*)(in_lds + abase + 32 * q + 16);
        const v8h m0 = *(const v8h*)(mask_h + mbase + 32 * q);
        const v8h m1 = *(const v8h*)(mask_h + mbase + 32 * q + 16);
        const v8h lo = i0 * m0;   // v_pk_mul_f16
        const v8h hi = i1 * m1;
        const v16h a = __builtin_shufflevector(
            lo, hi, 0, 1, 2, 3, 4, 5, 6, 7, 8, 9, 10, 11, 12, 13, 14, 15);
        const v16h br = *(const v16h*)(Bt + fr + (size_t)q * 512);
        const v16h bi = *(const v16h*)(Bt + fi + (size_t)q * 512);
        cr = wmma_f16(a, br, cr);
        ci = wmma_f16(a, bi, ci);
      }

      // D layout: lane (n, h), VGPR rr -> M = rr + 8h
      const int ff = wave * 2 + it;
#pragma unroll
      for (int rr = 0; rr < 8; ++rr) {
        const int m = rr + 8 * h;
        out_lds[(m * 16 + n) * 36 + ff] = cr[rr];
        out_lds[(m * 16 + n) * 36 + 16 + ff] = ci[rr];
      }
    }
    __syncthreads();

    // flush: per (b,l) two contiguous 64B channel runs
    for (int t2 = 0; t2 < 2; ++t2) {
      const int cid = tid * 2 + t2;  // 512 chunks of 16 floats
      const int m = cid >> 5;
      const int rest = cid & 31;
      const int nn = rest >> 1;
      const int comp = rest & 1;
      const float* src = &out_lds[(m * 16 + nn) * 36 + comp * 16];
      float* dst = out + ((size_t)(b0 + m) * 2048 + (l0 + nn)) * 128 +
                   comp * 64 + 16 * r;
#pragma unroll
      for (int v = 0; v < 4; ++v)
        *(float4*)(dst + 4 * v) = *(const float4*)(src + 4 * v);
    }
    __syncthreads();
  }
}

// ---------------------------------------------------------------------------
extern "C" void kernel_launch(void* const* d_in, const int* in_sizes, int n_in,
                              void* d_out, int out_size, void* d_ws, size_t ws_size,
                              hipStream_t stream) {
  const float* input = (const float*)d_in[0];  // (128, 2048, 2)
  const float* fc    = (const float*)d_in[1];
  const float* fbw   = (const float*)d_in[2];
  const float* tl    = (const float*)d_in[3];
  const float* st    = (const float*)d_in[4];
  float* out = (float*)d_out;                  // (128, 2048, 128)

  char* ws = (char*)d_ws;
  uint32_t* maskh2 = (uint32_t*)(ws);                         // 512 KB
  uint32_t* inh    = (uint32_t*)(ws + (512u << 10));          // 1 MB
  float*    kr     = (float*)(ws + (1536u << 10));            // 33 KB
  float*    ki     = (float*)(ws + (1536u << 10) + 33024);    // 33 KB
  _Float16* Bt     = (_Float16*)(ws + (1664u << 10));         // 1.125 MB

  sincnet_build_filters<<<64, 256, 0, stream>>>(fc, fbw, tl, st, maskh2, kr, ki);
  sincnet_cvt_input<<<1024, 256, 0, stream>>>(input, inh);
  sincnet_build_toeplitz<<<64, 256, 0, stream>>>(kr, ki, Bt);
  sincnet_wmma_main<<<dim3(128, 8), 256, 0, stream>>>(inh, maskh2, Bt, out);
}
